// AffinityImageEvent2_22445499089558
// MI455X (gfx1250) — matrix operand order repaired and checked by reading, compiled, not run on hardware
//
#include <hip/hip_runtime.h>
#include <stdint.h>

typedef _Float16 h16v __attribute__((ext_vector_type(16)));
typedef _Float16 h2v  __attribute__((ext_vector_type(2)));
typedef float    f8v  __attribute__((ext_vector_type(8)));

#define HH    256
#define WW    256
#define CCH   64
#define NBAT  4
#define ROWS  2                 // center rows per workgroup
#define XT    64                // center cols per workgroup
#define HR    (ROWS + 2)        // 4 halo rows
#define HWD   (XT + 2)          // 66 halo cols
#define NGRP  18                // 4-float groups staged per row (x0-4 .. x0+67)
#define RAWX  (NGRP * 4)        // 72
#define CHK   4                 // channels per async chunk
#define NCHUNK (CCH / CHK)      // 16
#define QP    36                // u32 pitch per pixel in Q tile (bank-conflict-free, 16B aligned)
#define NPX   (HR * HWD)        // 264 halo pixels
#define NTRF  (CHK * HR * NGRP) // 288 b128 transfers per chunk

#if __has_builtin(__builtin_amdgcn_s_wait_asynccnt)
#define WAIT_ASYNC(n) __builtin_amdgcn_s_wait_asynccnt(n)
#else
#define WAIT_ASYNC(n) asm volatile("s_wait_asynccnt %0" :: "i"(n) : "memory")
#endif

__global__ __launch_bounds__(256)
void aff_kernel(const float* __restrict__ img, const float* __restrict__ ev,
                float* __restrict__ out)
{
    __shared__ __align__(16) float    s_raw[2][CHK * HR * RAWX]; // raw f32 staging (double buf)
    __shared__ __align__(16) uint32_t s_q[NPX * QP];             // f16 img tile, [pixel][channel-pair]
    __shared__ float s_n2[NPX];                                  // per-pixel sum img^2
    __shared__ __align__(16) float s_dummy[4];

    const int tid   = threadIdx.x;
    const int xtile = blockIdx.x, ytile = blockIdx.y, b = blockIdx.z;
    const int x0 = xtile * XT, y0 = ytile * ROWS;
    const uint64_t imgbase = (uint64_t)(uintptr_t)img;

    // ---------------- Phase 1: async-stage raw img, build f16 Q tile + norms ----------
    auto issue = [&](int k, int p) {
#pragma unroll
        for (int it = 0; it < 2; ++it) {
            int idx = tid + it * 256;
            uint32_t ldso = 0, goff = 0;
            bool ok = false;
            if (idx < NTRF) {
                int c   = idx / (HR * NGRP);
                int rem = idx % (HR * NGRP);
                int row = rem / NGRP;
                int g   = rem % NGRP;
                int gy  = y0 - 1 + row;
                int gx  = x0 - 4 + g * 4;
                if (gy >= 0 && gy < HH && gx >= 0 && gx + 3 < WW) {
                    ok   = true;
                    ldso = (uint32_t)(uintptr_t)&s_raw[p][(c * HR + row) * RAWX + g * 4];
                    goff = (uint32_t)((((b * CCH + k * CHK + c) * HH + gy) * WW + gx) * 4);
                }
            }
            if (!ok) { ldso = (uint32_t)(uintptr_t)&s_dummy[0]; goff = 0; }
            // GVS mode: vdst(LDS byte addr), vaddr(32-bit byte offset), saddr(64-bit base)
            asm volatile("global_load_async_to_lds_b128 %0, %1, %2"
                         :: "v"(ldso), "v"(goff), "s"(imgbase) : "memory");
        }
    };

    float nacc0 = 0.f, nacc1 = 0.f;
    issue(0, 0);
    for (int k = 0; k < NCHUNK; ++k) {
        int p = k & 1;
        if (k + 1 < NCHUNK) {
            issue(k + 1, p ^ 1);
            WAIT_ASYNC(2);      // our 2 chunk-k ops done; chunk-(k+1)'s 2 still in flight
        } else {
            WAIT_ASYNC(0);
        }
        __syncthreads();        // all waves' chunk-k data visible in LDS
#pragma unroll
        for (int j = 0; j < 2; ++j) {
            int px = tid + j * 256;
            if (px < NPX) {
                int row = px / HWD, x = px % HWD;
                int gy = y0 - 1 + row, gx = x0 - 1 + x;
                bool inb = (gy >= 0) && (gy < HH) && (gx >= 0) && (gx < WW);
                float v0 = inb ? s_raw[p][(0 * HR + row) * RAWX + x + 3] : 0.f;
                float v1 = inb ? s_raw[p][(1 * HR + row) * RAWX + x + 3] : 0.f;
                float v2 = inb ? s_raw[p][(2 * HR + row) * RAWX + x + 3] : 0.f;
                float v3 = inb ? s_raw[p][(3 * HR + row) * RAWX + x + 3] : 0.f;
                float na = v0 * v0 + v1 * v1 + v2 * v2 + v3 * v3;
                if (j == 0) nacc0 += na; else nacc1 += na;
                h2v h01 = { (_Float16)v0, (_Float16)v1 };
                h2v h23 = { (_Float16)v2, (_Float16)v3 };
                uint32_t* qp = &s_q[px * QP + k * 2];
                qp[0] = __builtin_bit_cast(uint32_t, h01);
                qp[1] = __builtin_bit_cast(uint32_t, h23);
            }
        }
        __syncthreads();        // done reading raw buf before it is re-issued
    }
    if (tid < NPX)       s_n2[tid]       = nacc0;
    if (tid + 256 < NPX) s_n2[tid + 256] = nacc1;
    __syncthreads();

    // ---------------- Phase 2: WMMA correlation ----------------
    const int lane = tid & 31;
    const int wv   = tid >> 5;
    const int n    = lane & 15;
    const int hfl  = lane >> 4;
    // diag gather index: lane r reads lane r, lane r+8 reads lane r+24
    const int bprm = (((lane & 8) ? (lane & 7) + 24 : (lane & 7)) << 2);

    const int r  = wv >> 2;            // 0..1 : center row in tile
    const int xg = wv & 3;             // 0..3 : 16-pixel group
    const int y  = y0 + r;
    const int xl = xg * 16 + n;        // local center x (0..63)
    const int gx = x0 + xl;

    // A matrix: P = img(f16 from LDS) * ev(f16), 16 pixels x 64 channels; also Nev^2
    h16v a[2];
    float sev2 = 0.f;
    const int pixC = ((r + 1) * HWD + (xl + 1)) * QP;
#pragma unroll
    for (int t = 0; t < 2; ++t) {
        union { uint4 q[2]; h16v v; } ua;
        ua.q[0] = *reinterpret_cast<const uint4*>(&s_q[pixC + t * 16 + 4 * hfl]);
        ua.q[1] = *reinterpret_cast<const uint4*>(&s_q[pixC + t * 16 + 4 * hfl + 8]);
        h16v im = ua.v;
        h16v eh;
#pragma unroll
        for (int i = 0; i < 16; ++i) {
            int c = t * 32 + 8 * hfl + i + (i < 8 ? 0 : 8);  // A-layout channel for this lane
            float e = ev[((b * CCH + c) * HH + y) * WW + gx];
            sev2 += e * e;
            eh[i] = (_Float16)e;
        }
        a[t] = im * eh;
    }
    {   // pair-reduce Nev^2 across half-waves (lane ^ 16)
        int t16 = __builtin_amdgcn_ds_bpermute((lane ^ 16) << 2,
                                               __builtin_bit_cast(int, sev2));
        sev2 += __builtin_bit_cast(float, t16);
    }
    // denom: max(sqrt(x),1e-12)^2 == max(x,1e-24); use v_rsq/v_rcp (no div/sqrt slow paths)
    float cn2  = s_n2[(r + 1) * HWD + (xl + 1)];
    float cinv = __builtin_amdgcn_rsqf(fmaxf(cn2, 1e-24f)) *
                 __builtin_amdgcn_rsqf(fmaxf(sev2, 1e-24f));

    const int DY[8] = {-1, -1, -1,  0, 0,  1, 1, 1};
    const int DX[8] = {-1,  0,  1, -1, 1, -1, 0, 1};
#pragma unroll
    for (int o = 0; o < 8; ++o) {
        int rq   = r + 1 + DY[o];
        int xq   = xl + 1 + DX[o];
        int pixB = (rq * HWD + xq) * QP;
        f8v acc  = {0.f, 0.f, 0.f, 0.f, 0.f, 0.f, 0.f, 0.f};
#pragma unroll
        for (int t = 0; t < 2; ++t) {
            union { uint4 q[2]; h16v v; } ub;
            ub.q[0] = *reinterpret_cast<const uint4*>(&s_q[pixB + t * 16 + 4 * hfl]);
            ub.q[1] = *reinterpret_cast<const uint4*>(&s_q[pixB + t * 16 + 4 * hfl + 8]);
            h16v bq = ub.v;
            bq = bq * bq;   // Q = img^2 at neighbor (v_pk_mul_f16)
            acc = __builtin_amdgcn_wmma_f32_16x16x32_f16(false, a[t], false, bq,
                                                         (short)0, acc, false, false);
        }
        // extract diagonal of D: VGPR rr holds diag at lanes rr and rr+24
        float diag = 0.f;
#pragma unroll
        for (int rr = 0; rr < 8; ++rr) {
            int tb = __builtin_amdgcn_ds_bpermute(bprm,
                                                  __builtin_bit_cast(int, acc[rr]));
            if ((lane & 7) == rr) diag = __builtin_bit_cast(float, tb);
        }
        float nb2     = s_n2[rq * HWD + xq];
        float inv_nb2 = __builtin_amdgcn_rcpf(fmaxf(nb2, 1e-24f));
        float val     = fmaxf(diag * cinv * inv_nb2, 0.f);
        if (lane < 16)
            out[((b * 8 + o) * HH + y) * WW + gx] = val;
    }
}

extern "C" void kernel_launch(void* const* d_in, const int* in_sizes, int n_in,
                              void* d_out, int out_size, void* d_ws, size_t ws_size,
                              hipStream_t stream) {
    (void)in_sizes; (void)n_in; (void)out_size; (void)d_ws; (void)ws_size;
    const float* img = (const float*)d_in[0];
    const float* ev  = (const float*)d_in[1];
    float* out = (float*)d_out;
    dim3 grid(WW / XT, HH / ROWS, NBAT);   // (4, 128, 4)
    aff_kernel<<<grid, dim3(256), 0, stream>>>(img, ev, out);
}
